// NNPytorchAD_12610023981214
// MI455X (gfx1250) — compile-verified, matrix operand order checked by reading, not run on hardware
//
#include <hip/hip_runtime.h>
#include <hip/hip_bf16.h>

typedef __bf16 bf16_t;
typedef __attribute__((ext_vector_type(16))) __bf16 v16bf;
typedef __attribute__((ext_vector_type(8)))  __bf16 v8bf;
typedef __attribute__((ext_vector_type(8)))  float  v8f;

#define B_SZ   1024
#define D_IN   64
#define HID    128
#define D_OUT  16
#define NB     4          // batches per block
#define NTILE  256        // d2f M-tiles (4096 rows / 16)
#define NWAVE  8

// ---------------------------------------------------------------------------
// Kernel 1: precompute U[(k*64+i), h] = W1[k,h]*W1[i,h]  (bf16, 4096x128)
//           and a bf16 copy of W1 (64x128). Both K-contiguous for WMMA A loads.
// ---------------------------------------------------------------------------
__global__ __launch_bounds__(256) void precompute_U(const float* __restrict__ W1,
                                                    bf16_t* __restrict__ U,
                                                    bf16_t* __restrict__ W1b) {
  int idx = blockIdx.x * blockDim.x + threadIdx.x;   // 2048 blocks * 256 = 524288
  if (idx < D_IN * D_IN * HID) {
    int h   = idx & (HID - 1);
    int row = idx >> 7;
    int i   = row & (D_IN - 1);
    int k   = row >> 6;
    U[idx] = (bf16_t)(W1[k * HID + h] * W1[i * HID + h]);
  }
  if (idx < D_IN * HID) {
    W1b[idx] = (bf16_t)W1[idx];
  }
}

// ---------------------------------------------------------------------------
// Kernel 2: one block per NB=4 batch elements. 256 threads = 8 waves (wave32).
//  Stage:   per batch: z = x@W1+b1, a=tanh(z), s=1-a^2, t=-2*a*s ; f to global;
//           Gs[nb][j][h]=s*W2[h,j], Gt[nb][j][h]=t*W2[h,j] bf16 in LDS,
//           K-contiguous per column j -> B fragments via ds_load_b128.
//  GEMMs:   each A-fragment load (U tile rows, L2-resident) feeds 4 independent
//           WMMA accumulation chains (one per batch) -> 16 WMMAs per 8 loads.
//           Tile loop has a compile-time trip count (32/wave) -> uniform branch.
// ---------------------------------------------------------------------------
__global__ __launch_bounds__(256) void wmma_grads(const float*  __restrict__ x,
                                                  const float*  __restrict__ W1,
                                                  const float*  __restrict__ b1,
                                                  const float*  __restrict__ W2,
                                                  const float*  __restrict__ b2,
                                                  const bf16_t* __restrict__ U,
                                                  const bf16_t* __restrict__ W1b,
                                                  float* __restrict__ f_out,
                                                  float* __restrict__ df_out,
                                                  float* __restrict__ d2f_out) {
  __shared__ float  xs[NB * D_IN];
  __shared__ float  as[NB][HID];
  __shared__ bf16_t Gs[NB][D_OUT][HID];   // s[h]*W2[h,j], K-contiguous per j
  __shared__ bf16_t Gt[NB][D_OUT][HID];   // t[h]*W2[h,j]

  const int tid  = threadIdx.x;
  const int b0   = blockIdx.x * NB;
  const int lane = tid & 31;
  const int wave = tid >> 5;           // 0..7
  const int n    = lane & 15;          // output column N / A row M
  const int hi   = lane >> 4;          // lane half (0/1)
  const int kh   = hi ? 8 : 0;         // K offset of lane half in A/B layout

  // ---- stage: activations + B matrices for NB batches ----
  xs[tid] = x[(size_t)b0 * D_IN + tid];            // 256 == NB*D_IN
  __syncthreads();

  {
    const int h   = tid & (HID - 1);
    const int nb0 = tid >> 7;                      // 0 or 1
#pragma unroll
    for (int it = 0; it < 2; ++it) {
      const int nb = nb0 + 2 * it;                 // {0,2} and {1,3}
      float z = b1[h];
#pragma unroll
      for (int i = 0; i < D_IN; ++i) z += xs[nb * D_IN + i] * W1[i * HID + h];
      float a = tanhf(z);
      as[nb][h] = a;
      float s = 1.0f - a * a;
      float t = -2.0f * a * s;
#pragma unroll
      for (int j = 0; j < D_OUT; ++j) {
        float w2 = W2[h * D_OUT + j];
        Gs[nb][j][h] = (bf16_t)(s * w2);
        Gt[nb][j][h] = (bf16_t)(t * w2);
      }
    }
  }
  __syncthreads();

  if (tid < NB * D_OUT) {
    const int nb = tid >> 4, j = tid & 15;
    float acc = b2[j];
#pragma unroll
    for (int h = 0; h < HID; ++h) acc += as[nb][h] * W2[h * D_OUT + j];
    f_out[(size_t)(b0 + nb) * D_OUT + j] = acc;
  }

  // ---- B fragments (Gt) for all NB batches: 4 K-chunks of 32 each ----
  v16bf bt[NB][4];
#pragma unroll
  for (int nb = 0; nb < NB; ++nb) {
#pragma unroll
    for (int c = 0; c < 4; ++c) {
      const int k0 = c * 32;
      const v8bf lo = *(const v8bf*)(&Gt[nb][n][k0 + kh]);
      const v8bf hc = *(const v8bf*)(&Gt[nb][n][k0 + 16 + kh]);
#pragma unroll
      for (int e = 0; e < 8; ++e) { bt[nb][c][e] = lo[e]; bt[nb][c][e + 8] = hc[e]; }
    }
  }

  // ---- d2f: 256 M-tiles of 16 rows; 32 per wave, constant trip count ----
  const bf16_t* arow0 = U + (size_t)(wave * 16 + n) * HID;
  float* dst0 = d2f_out + (size_t)b0 * (D_IN * D_IN * D_OUT)
              + (size_t)(wave * 16 + 8 * hi) * D_OUT + n;
  for (int it = 0; it < NTILE / NWAVE; ++it) {
    const bf16_t* arow = arow0 + (size_t)it * (NWAVE * 16 * HID);
    v16bf a[4];
#pragma unroll
    for (int c = 0; c < 4; ++c) {
      const int k0 = c * 32;
      const v8bf lo = *(const v8bf*)(arow + k0 + kh);
      const v8bf hc = *(const v8bf*)(arow + k0 + 16 + kh);
#pragma unroll
      for (int e = 0; e < 8; ++e) { a[c][e] = lo[e]; a[c][e + 8] = hc[e]; }
    }
#pragma unroll
    for (int nb = 0; nb < NB; ++nb) {
      v8f acc = {};
#pragma unroll
      for (int c = 0; c < 4; ++c) {
        acc = __builtin_amdgcn_wmma_f32_16x16x32_bf16(
                  false, a[c], false, bt[nb][c], (short)0, acc, false, false);
      }
      // C/D layout: VGPR v -> row M = v + 8*hi, lane n -> col N
      float* dst = dst0 + (size_t)nb * (D_IN * D_IN * D_OUT)
                 + (size_t)it * (NWAVE * 16 * D_OUT);
#pragma unroll
      for (int v = 0; v < 8; ++v) dst[v * D_OUT] = acc[v];
    }
  }

  // ---- df: 4 M-tiles (rows of W1), one per wave on waves 0..3 ----
  if (wave < 4) {
    const bf16_t* arow = W1b + (size_t)(wave * 16 + n) * HID;
    v16bf a[4];
#pragma unroll
    for (int c = 0; c < 4; ++c) {
      const int k0 = c * 32;
      const v8bf lo = *(const v8bf*)(arow + k0 + kh);
      const v8bf hc = *(const v8bf*)(arow + k0 + 16 + kh);
#pragma unroll
      for (int e = 0; e < 8; ++e) { a[c][e] = lo[e]; a[c][e + 8] = hc[e]; }
    }
#pragma unroll
    for (int nb = 0; nb < NB; ++nb) {
      v8f acc = {};
#pragma unroll
      for (int c = 0; c < 4; ++c) {
        const int k0 = c * 32;
        const v8bf lo = *(const v8bf*)(&Gs[nb][n][k0 + kh]);
        const v8bf hc = *(const v8bf*)(&Gs[nb][n][k0 + 16 + kh]);
        v16bf bsc;
#pragma unroll
        for (int e = 0; e < 8; ++e) { bsc[e] = lo[e]; bsc[e + 8] = hc[e]; }
        acc = __builtin_amdgcn_wmma_f32_16x16x32_bf16(
                  false, a[c], false, bsc, (short)0, acc, false, false);
      }
      float* dst = df_out + (size_t)(b0 + nb) * (D_IN * D_OUT)
                 + (size_t)(wave * 16 + 8 * hi) * D_OUT + n;
#pragma unroll
      for (int v = 0; v < 8; ++v) dst[v * D_OUT] = acc[v];
    }
  }
}

// ---------------------------------------------------------------------------
extern "C" void kernel_launch(void* const* d_in, const int* in_sizes, int n_in,
                              void* d_out, int out_size, void* d_ws, size_t ws_size,
                              hipStream_t stream) {
  const float* x  = (const float*)d_in[0];
  const float* W1 = (const float*)d_in[1];
  const float* b1 = (const float*)d_in[2];
  const float* W2 = (const float*)d_in[3];
  const float* b2 = (const float*)d_in[4];

  float* f_out   = (float*)d_out;                       // [1024,16]
  float* df_out  = f_out  + (size_t)B_SZ * D_OUT;       // [1024,64,16]
  float* d2f_out = df_out + (size_t)B_SZ * D_IN * D_OUT;// [1024,64,64,16]

  char*   ws  = (char*)d_ws;
  bf16_t* U   = (bf16_t*)ws;                                   // 4096*128 bf16 = 1MB
  bf16_t* W1b = (bf16_t*)(ws + (size_t)D_IN * D_IN * HID * 2); // 64*128 bf16 = 16KB

  precompute_U<<<(D_IN * D_IN * HID + 255) / 256, 256, 0, stream>>>(W1, U, W1b);
  wmma_grads<<<B_SZ / NB, 256, 0, stream>>>(x, W1, b1, W2, b2, U, W1b,
                                            f_out, df_out, d2f_out);
}
